// GCN_43207370998079
// MI455X (gfx1250) — compile-verified, matrix operand order checked by reading, not run on hardware
//
#include <hip/hip_runtime.h>
#include <hip/hip_bf16.h>

// ---------------------------------------------------------------------------
// GCN forward for MI455X (gfx1250, wave32, WMMA + TDM).
//   xw1 = x @ W1                     (bf16x3 compensated WMMA, fp32-accurate)
//   agg1 = scatter_add(edge_w * xw1[src] -> dst)   (fp32 HW atomics)
//   hw2 = relu(agg1 + b1) @ W2       (exact f32 WMMA 16x16x4, fused epilogue)
//   out  = scatter_add(edge_w * hw2[src] -> dst) + b2
// W1 k-tiles are DMA'd to LDS by the Tensor Data Mover (tensor_load_to_lds),
// then converted ONCE to packed/transposed bf16 hi/lo tiles for ds-only
// B-fragment builds.
// ---------------------------------------------------------------------------

#define NN 100000      // nodes
#define NE 800000      // edges
#define NF 500         // input features
#define NH 64          // hidden
#define NC 16          // classes
#define WST 34         // padded LDS stride (halves) for packed W tiles

typedef __attribute__((ext_vector_type(16))) __bf16 v16bf;
typedef __attribute__((ext_vector_type(8)))  float  v8f;
typedef __attribute__((ext_vector_type(2)))  float  v2f;

#if __has_builtin(__builtin_amdgcn_tensor_load_to_lds) && \
    __has_builtin(__builtin_amdgcn_s_wait_tensorcnt)
#define HAVE_TDM 1
typedef unsigned u32x4 __attribute__((ext_vector_type(4)));
typedef int      i32x8 __attribute__((ext_vector_type(8)));
typedef int      i32x4 __attribute__((ext_vector_type(4)));
#endif

// ---------------------------------------------------------------------------
// GEMM1: xw1[NN,64] = x[NN,500] @ W1[500,64], bf16x3 (hi*hi + hi*lo + lo*hi)
// Block = 256 threads = 8 waves; wave -> 16x64 output tile; block -> 128 rows.
// ---------------------------------------------------------------------------
__global__ __launch_bounds__(256) void gcn_gemm1(const float* __restrict__ x,
                                                 const float* __restrict__ w1,
                                                 float* __restrict__ xw1) {
  __shared__ float  wraw[32 * NH];    // 8 KB raw f32 tile W1[kt:kt+32, 0:64]
  __shared__ __bf16 whi[NH * WST];    // packed hi, transposed [n][k], padded
  __shared__ __bf16 wlo[NH * WST];    // packed lo residual

  const int lane = threadIdx.x & 31;
  const int wv   = threadIdx.x >> 5;
  const int half = lane >> 4;        // K-half select per ISA A/B layout
  const int l15  = lane & 15;
  const int m0   = (blockIdx.x * 8 + wv) * 16;
  const int mrow = m0 + l15;
  const float* __restrict__ arow = x + (size_t)(mrow < NN ? mrow : 0) * NF;

  v8f acc[4] = {};   // 4 x (16x16 f32) accumulators covering N=64

  for (int kt = 0; kt < NF; kt += 32) {
#if HAVE_TDM
    // --- Tensor Data Mover: DMA W1[kt:kt+32, 0:64] (f32) into LDS ---------
    if (threadIdx.x < 32) {          // wave 0 issues; TDM is one op per wave
      const unsigned lds = (unsigned)(size_t)(void*)wraw;
      const unsigned long long ga =
          (unsigned long long)(size_t)(w1 + (size_t)kt * NH);
      // D# group0: count=1 | lds_addr | global_addr[56:0] | type=2
      u32x4 g0 = { 1u, lds, (unsigned)ga,
                   (unsigned)((ga >> 32) & 0x1FFFFFFu) | 0x80000000u };
      // D# group1: data_size=4B; tensor_dim0=64; tensor_dim1=NF-kt (OOB rows
      // beyond the tensor read back as zero -> pads the k>=500 tail);
      // tile_dim0=64, tile_dim1=32; tensor_dim0_stride=64.
      const int dim1 = NF - kt;
      i32x8 g1 = { 0x00020000, (NH << 16), ((dim1 & 0xffff) << 16),
                   (NH << 16), 32, NH, 0, 0 };
      i32x4 gz4 = { 0, 0, 0, 0 };
      i32x8 gz8 = { 0, 0, 0, 0, 0, 0, 0, 0 };
      // 6-arg toolchain variant (clang-23 / therock-10.0 headers)
      __builtin_amdgcn_tensor_load_to_lds(g0, g1, gz4, gz4, gz8, 0);
      __builtin_amdgcn_s_wait_tensorcnt(0);
    }
#else
    // --- fallback: cooperative staging of the W1 tile (zero-pad K>=500) ---
    for (int i = threadIdx.x; i < 32 * NH; i += 256) {
      int kk = kt + (i >> 6);
      wraw[i] = (kk < NF) ? w1[kk * NH + (i & (NH - 1))] : 0.0f;
    }
#endif
    __syncthreads();

    // --- one-time convert: raw f32 tile -> packed/transposed bf16 hi/lo ---
    for (int i = threadIdx.x; i < 32 * NH; i += 256) {
      int n = i & (NH - 1);
      int k = i >> 6;
      float  f = wraw[k * NH + n];
      __bf16 h = (__bf16)f;                    // native RNE cvt on gfx1250
      whi[n * WST + k] = h;
      wlo[n * WST + k] = (__bf16)(f - (float)h);
    }
    __syncthreads();

    // prefetch next A segment while we compute (speculative, safe)
    __builtin_prefetch(arow + kt + 32, 0, 0);

    // --- A fragment: 16-bit A 16x32 layout, split hi/lo for bf16x3 ---
    v16bf ahi, alo;
#pragma unroll
    for (int i = 0; i < 16; ++i) {
      int v  = i >> 1;
      int kb = (v < 4) ? (v << 1) : (16 + ((v - 4) << 1));
      int k  = kt + kb + (i & 1) + half * 8;
      float  f = (k < NF) ? arow[k] : 0.0f;
      __bf16 h = (__bf16)f;
      ahi[i] = h;
      alo[i] = (__bf16)(f - (float)h);
    }

#pragma unroll
    for (int nt = 0; nt < 4; ++nt) {
      const int ncol = nt * 16 + l15;
      const __bf16* __restrict__ ph = whi + ncol * WST + half * 8;
      const __bf16* __restrict__ pl = wlo + ncol * WST + half * 8;
      v16bf bhi, blo;
#pragma unroll
      for (int v = 0; v < 8; ++v) {
        int kl = (v < 4) ? (v << 1) : (16 + ((v - 4) << 1));
        bhi[2 * v]     = ph[kl];
        bhi[2 * v + 1] = ph[kl + 1];
        blo[2 * v]     = pl[kl];
        blo[2 * v + 1] = pl[kl + 1];
      }
      // bf16x3: hi*hi + hi*lo + lo*hi  (drop lo*lo) -> ~fp32 accuracy
      acc[nt] = __builtin_amdgcn_wmma_f32_16x16x32_bf16(
          false, ahi, false, bhi, (short)0, acc[nt], false, false);
      acc[nt] = __builtin_amdgcn_wmma_f32_16x16x32_bf16(
          false, ahi, false, blo, (short)0, acc[nt], false, false);
      acc[nt] = __builtin_amdgcn_wmma_f32_16x16x32_bf16(
          false, alo, false, bhi, (short)0, acc[nt], false, false);
    }
    __syncthreads();   // all fragment reads done before next tile overwrite
  }

  // --- store: f32 C/D layout (VGPR r: lanes0-15 M=r, lanes16-31 M=r+8) ---
#pragma unroll
  for (int nt = 0; nt < 4; ++nt) {
    const int col = nt * 16 + l15;
#pragma unroll
    for (int r = 0; r < 8; ++r) {
      int row = m0 + half * 8 + r;
      if (row < NN) xw1[(size_t)row * NH + col] = acc[nt][r];
    }
  }
}

// ---------------------------------------------------------------------------
// GEMM2 fused: hw2[NN,16] = relu(agg1 + b1) @ W2[64,16], exact f32 WMMA.
// Wave -> 16x16 tile; K=64 as 16 steps of 16x16x4.
// ---------------------------------------------------------------------------
__global__ __launch_bounds__(256) void gcn_gemm2(const float* __restrict__ agg,
                                                 const float* __restrict__ b1,
                                                 const float* __restrict__ w2,
                                                 float* __restrict__ hw2) {
  const int lane = threadIdx.x & 31;
  const int wv   = threadIdx.x >> 5;
  const int half = lane >> 4;
  const int l15  = lane & 15;
  const int m0   = (blockIdx.x * 8 + wv) * 16;
  const int mrow = m0 + l15;
  const float* __restrict__ arow = agg + (size_t)(mrow < NN ? mrow : 0) * NH;

  v8f acc = {};
  for (int kt = 0; kt < NH; kt += 4) {
    v2f a, b;
#pragma unroll
    for (int v = 0; v < 2; ++v) {
      int k = kt + half * 2 + v;          // f32 A 16x4 layout
      float hv = arow[k] + b1[k];         // fused bias
      a[v] = hv > 0.0f ? hv : 0.0f;       // fused ReLU
      b[v] = w2[k * NC + l15];            // f32 B 4x16 layout (mirrored)
    }
    acc = __builtin_amdgcn_wmma_f32_16x16x4_f32(
        false, a, false, b, (short)0, acc, false, false);
  }

#pragma unroll
  for (int r = 0; r < 8; ++r) {
    int row = m0 + half * 8 + r;
    if (row < NN) hw2[(size_t)row * NC + l15] = acc[r];
  }
}

// ---------------------------------------------------------------------------
// SpMM scatter: out[dst[e], f] += edge_w[e] * feat[src[e], f]
// One lane per (edge, feature): coalesced feature-row gather, HW fp32 atomics.
// ---------------------------------------------------------------------------
template <int LOGF>
__global__ __launch_bounds__(256) void gcn_spmm(const float* __restrict__ feat,
                                                const float* __restrict__ ew,
                                                const int* __restrict__ src,
                                                const int* __restrict__ dst,
                                                float* __restrict__ out) {
  const unsigned long gid = (unsigned long)blockIdx.x * 256u + threadIdx.x;
  const int e = (int)(gid >> LOGF);
  const int f = (int)(gid & ((1u << LOGF) - 1u));
  if (e >= NE) return;
  // stream the edge arrays ahead of the wavefronts (global_prefetch_b8)
  __builtin_prefetch(src + e + 4096, 0, 0);
  __builtin_prefetch(dst + e + 4096, 0, 0);
  __builtin_prefetch(ew  + e + 4096, 0, 0);
  const float w = ew[e];
  const int s = src[e];
  const int d = dst[e];
  const float v = w * feat[((size_t)s << LOGF) | f];
  unsafeAtomicAdd(&out[((size_t)d << LOGF) | f], v);  // global_atomic_add_f32
}

// out[i, c] = b2[c]  (bias pre-fill before final scatter)
__global__ __launch_bounds__(256) void gcn_init_out(float* __restrict__ out,
                                                    const float* __restrict__ b2) {
  const unsigned gid = blockIdx.x * 256u + threadIdx.x;
  if (gid < (unsigned)(NN * NC)) out[gid] = b2[gid & (NC - 1)];
}

extern "C" void kernel_launch(void* const* d_in, const int* in_sizes, int n_in,
                              void* d_out, int out_size, void* d_ws, size_t ws_size,
                              hipStream_t stream) {
  const float* x   = (const float*)d_in[0];
  const float* W1  = (const float*)d_in[1];
  const float* b1  = (const float*)d_in[2];
  const float* W2  = (const float*)d_in[3];
  const float* b2  = (const float*)d_in[4];
  const float* ew  = (const float*)d_in[5];
  const int*   src = (const int*)d_in[6];
  const int*   dst = (const int*)d_in[7];
  float* out = (float*)d_out;

  float* xw1  = (float*)d_ws;                    // [NN,64] 25.6 MB
  float* agg1 = xw1  + (size_t)NN * NH;          // [NN,64] 25.6 MB
  float* hw2  = agg1 + (size_t)NN * NH;          // [NN,16]  6.4 MB

  const dim3 blk(256);
  const int mblocks = (NN + 127) / 128;          // 782 (8 waves x 16 rows)

  (void)hipMemsetAsync(agg1, 0, sizeof(float) * (size_t)NN * NH, stream);
  gcn_gemm1<<<mblocks, blk, 0, stream>>>(x, W1, xw1);
  gcn_spmm<6><<<(NE * 64) / 256, blk, 0, stream>>>(xw1, ew, src, dst, agg1);
  gcn_gemm2<<<mblocks, blk, 0, stream>>>(agg1, b1, W2, hw2);
  gcn_init_out<<<(NN * NC + 255) / 256, blk, 0, stream>>>(out, b2);
  gcn_spmm<4><<<(NE * 16) / 256, blk, 0, stream>>>(hw2, ew, src, dst, out);
}
